// GRU_6975026889166
// MI455X (gfx1250) — compile-verified, matrix operand order checked by reading, not run on hardware
//
#include <hip/hip_runtime.h>

// GRU: B=128, S=512, I=1024, H=1024, G=3H=3072
#define Bn 128
#define Sn 512
#define In 1024
#define Hn 1024
#define Gn 3072

typedef __attribute__((ext_vector_type(16))) __bf16 v16bf;
typedef __attribute__((ext_vector_type(8)))  __bf16 v8bf;
typedef __attribute__((ext_vector_type(4)))  __bf16 v4bf;
typedef __attribute__((ext_vector_type(8)))  float  v8f;

__device__ __forceinline__ float sigmoidf_fast(float x) {
    return 1.0f / (1.0f + __expf(-x));
}
__device__ __forceinline__ float tanhf_fast(float x) {
    return 1.0f - 2.0f / (__expf(2.0f * x) + 1.0f);
}

// ---------------------------------------------------------------------------
// Pack a (3072 x 1024) f32 weight matrix into bf16 B-fragment order:
// frag f = nt*32 + kt; per lane l (0..31), elem e (0..15):
//   n = nt*16 + (l&15), half = l>>4
//   e<8 : k = kt*32 + half*8 + e
//   e>=8: k = kt*32 + 16 + half*8 + (e-8)
// A wave's fragment for (nt,kt) is 32 lanes x 32B contiguous (1KB).
// ---------------------------------------------------------------------------
__global__ void pack_w_kernel(const float* __restrict__ W, __bf16* __restrict__ out) {
    int idx = blockIdx.x * 256 + threadIdx.x;   // < 3072*1024
    int e  = idx & 15;
    int l  = (idx >> 4) & 31;
    int kt = (idx >> 9) & 31;
    int nt = idx >> 14;
    int n    = nt * 16 + (l & 15);
    int half = l >> 4;
    int k = kt * 32 + ((e < 8) ? (half * 8 + e) : (16 + half * 8 + (e - 8)));
    out[idx] = (__bf16)W[n * 1024 + k];
}

// Pack h0 (B x H f32) into the bf16 A-fragment layout:
// frag index ((mt*32 + kt)*32 + lane)*16 + e ;  row = mt*16 + (lane&15)
__global__ void pack_h0_kernel(const float* __restrict__ h0, __bf16* __restrict__ out) {
    int idx = blockIdx.x * 256 + threadIdx.x;   // < 128*1024
    int e  = idx & 15;
    int l  = (idx >> 4) & 31;
    int kt = (idx >> 9) & 31;
    int mt = idx >> 14;
    int row  = mt * 16 + (l & 15);
    int half = l >> 4;
    int k = kt * 32 + ((e < 8) ? (half * 8 + e) : (16 + half * 8 + (e - 8)));
    out[idx] = (__bf16)h0[row * Hn + k];
}

// x (B,S,I) f32 -> bf16 row-major (rows are A-matrix rows, m = b*S+s)
__global__ void cvt_x_kernel(const float4* __restrict__ x, v4bf* __restrict__ xb) {
    size_t i = (size_t)blockIdx.x * 256 + threadIdx.x;
    float4 f = x[i];
    v4bf b;
    b[0] = (__bf16)f.x; b[1] = (__bf16)f.y; b[2] = (__bf16)f.z; b[3] = (__bf16)f.w;
    xb[i] = b;
}

// hn output = h0 (reference returns h0[None].copy())
__global__ void copy_hn_kernel(const float* __restrict__ h0, float* __restrict__ dst) {
    int i = blockIdx.x * 256 + threadIdx.x;
    dst[i] = h0[i];
}

// ---------------------------------------------------------------------------
// Phase A: gx[m,g] = sum_i x[m,i]*W_ih[g,i] + bias[g];  M=65536, N=3072, K=1024
// 256 threads = 8 waves; block tile 32(M) x 256(N); wave = 16x64 (4 tiles).
// Register double-buffered over kt so prefetch loads stay outstanding
// during the WMMAs (partial s_wait_loadcnt instead of drain-to-0).
// ---------------------------------------------------------------------------
__global__ __launch_bounds__(256) void gemm_gx_kernel(
    const __bf16* __restrict__ xb, const __bf16* __restrict__ wihp,
    const float* __restrict__ bias, float* __restrict__ gx) {
    int tid  = threadIdx.x;
    int wave = tid >> 5, lane = tid & 31;
    int mw = wave >> 2, nw = wave & 3;
    int m0 = blockIdx.x * 32 + mw * 16;
    int n0 = blockIdx.y * 256 + nw * 64;

    int row  = m0 + (lane & 15);
    int half = lane >> 4;
    const __bf16* xrow = xb + (size_t)row * In;
    const v16bf* Wv = (const v16bf*)wihp;
    int ntbase = n0 >> 4;

    v8f c0 = {}, c1 = {}, c2 = {}, c3 = {};

    // prologue: fragments for kt = 0
    v16bf a;
    {
        v8bf alo = *(const v8bf*)(xrow + half * 8);
        v8bf ahi = *(const v8bf*)(xrow + 16 + half * 8);
        #pragma unroll
        for (int i = 0; i < 8; ++i) { a[i] = alo[i]; a[i + 8] = ahi[i]; }
    }
    size_t bb0 = (size_t)ntbase * 32 * 32 + lane;
    v16bf b0 = Wv[bb0];
    v16bf b1 = Wv[bb0 + 1024];
    v16bf b2 = Wv[bb0 + 2048];
    v16bf b3 = Wv[bb0 + 3072];

    for (int kt = 0; kt < In / 32 - 1; ++kt) {
        // prefetch kt+1
        int kb = (kt + 1) * 32;
        v8bf alo = *(const v8bf*)(xrow + kb + half * 8);
        v8bf ahi = *(const v8bf*)(xrow + kb + 16 + half * 8);
        size_t bb = ((size_t)ntbase * 32 + kt + 1) * 32 + lane;
        v16bf nb0 = Wv[bb];
        v16bf nb1 = Wv[bb + 1024];
        v16bf nb2 = Wv[bb + 2048];
        v16bf nb3 = Wv[bb + 3072];

        // math on kt
        c0 = __builtin_amdgcn_wmma_f32_16x16x32_bf16(false, a, false, b0, (short)0, c0, false, false);
        c1 = __builtin_amdgcn_wmma_f32_16x16x32_bf16(false, a, false, b1, (short)0, c1, false, false);
        c2 = __builtin_amdgcn_wmma_f32_16x16x32_bf16(false, a, false, b2, (short)0, c2, false, false);
        c3 = __builtin_amdgcn_wmma_f32_16x16x32_bf16(false, a, false, b3, (short)0, c3, false, false);

        #pragma unroll
        for (int i = 0; i < 8; ++i) { a[i] = alo[i]; a[i + 8] = ahi[i]; }
        b0 = nb0; b1 = nb1; b2 = nb2; b3 = nb3;
    }
    c0 = __builtin_amdgcn_wmma_f32_16x16x32_bf16(false, a, false, b0, (short)0, c0, false, false);
    c1 = __builtin_amdgcn_wmma_f32_16x16x32_bf16(false, a, false, b1, (short)0, c1, false, false);
    c2 = __builtin_amdgcn_wmma_f32_16x16x32_bf16(false, a, false, b2, (short)0, c2, false, false);
    c3 = __builtin_amdgcn_wmma_f32_16x16x32_bf16(false, a, false, b3, (short)0, c3, false, false);

    // C/D layout: lane holds column n = lane&15; VGPR v -> row (lane>>4)*8 + v
    int nl = lane & 15;
    int mb = (lane >> 4) * 8;
    #pragma unroll
    for (int j = 0; j < 4; ++j) {
        v8f c = (j == 0) ? c0 : (j == 1) ? c1 : (j == 2) ? c2 : c3;
        int   ncol = n0 + j * 16 + nl;
        float bv   = bias[ncol];
        #pragma unroll
        for (int v = 0; v < 8; ++v) {
            int mrow = m0 + mb + v;
            gx[(size_t)mrow * Gn + ncol] = c[v] + bv;
        }
    }
}

// ---------------------------------------------------------------------------
// Phase B step: gh = h_prev @ W_hh^T, fused gates, h update.
// h_prev comes in PRE-PACKED bf16 A-fragment layout (hpin, 256KB, L2-resident):
// no LDS staging, no conversion, no barrier on the scan critical path.
// Epilogue operands (gx gates + f32 h_prev) are prefetched (global_prefetch_b8)
// BEFORE the 96-WMMA K-loop so their latency is hidden under the matrix math.
// Epilogue writes out[:,t,:] (f32) AND h_next re-packed (bf16) for step t+1.
// ---------------------------------------------------------------------------
__global__ __launch_bounds__(128) void gru_step_kernel(
    int t, const float* __restrict__ gx, const __bf16* __restrict__ whhp,
    const float* __restrict__ bias, const float* __restrict__ hprevf,
    long hstride, const v16bf* __restrict__ hpin, __bf16* __restrict__ hpout,
    float* __restrict__ out) {
    int tid  = threadIdx.x;
    int mt   = blockIdx.x;                    // m-tile 0..7, m0 = mt*16
    int m0   = mt * 16;
    int nh   = blockIdx.y * 64 + (tid >> 5) * 16;
    int lane = tid & 31;

    int nl = lane & 15;
    int mb = (lane >> 4) * 8;
    int ncol = nh + nl;

    // ---- prefetch epilogue operands; latency hidden by the K-loop ----
    #pragma unroll
    for (int v = 0; v < 8; ++v) {
        int brow = m0 + mb + v;
        size_t g = ((size_t)brow * Sn + t) * Gn;
        __builtin_prefetch(&gx[g + ncol], 0, 3);
        __builtin_prefetch(&gx[g + Hn + ncol], 0, 3);
        __builtin_prefetch(&gx[g + 2 * Hn + ncol], 0, 3);
        __builtin_prefetch(&hprevf[(size_t)brow * hstride + ncol], 0, 3);
    }

    const v16bf* Wv = (const v16bf*)whhp;
    size_t abase = ((size_t)mt * 32) * 32 + lane;          // (mt*32 + kt)*32 + lane
    size_t bzb = ((size_t)((0 * Hn + nh) >> 4) * 32) * 32 + lane;
    size_t brb = ((size_t)((1 * Hn + nh) >> 4) * 32) * 32 + lane;
    size_t bnb = ((size_t)((2 * Hn + nh) >> 4) * 32) * 32 + lane;

    v8f cz = {}, cr = {}, cn = {};

    // prologue (kt = 0)
    v16bf a  = hpin[abase];
    v16bf bz = Wv[bzb];
    v16bf br = Wv[brb];
    v16bf bn = Wv[bnb];

    for (int kt = 0; kt < Hn / 32 - 1; ++kt) {
        int o = (kt + 1) * 32;
        v16bf an  = hpin[abase + o];
        v16bf bzn = Wv[bzb + o];
        v16bf brn = Wv[brb + o];
        v16bf bnn = Wv[bnb + o];

        cz = __builtin_amdgcn_wmma_f32_16x16x32_bf16(false, a, false, bz, (short)0, cz, false, false);
        cr = __builtin_amdgcn_wmma_f32_16x16x32_bf16(false, a, false, br, (short)0, cr, false, false);
        cn = __builtin_amdgcn_wmma_f32_16x16x32_bf16(false, a, false, bn, (short)0, cn, false, false);

        a = an; bz = bzn; br = brn; bn = bnn;
    }
    cz = __builtin_amdgcn_wmma_f32_16x16x32_bf16(false, a, false, bz, (short)0, cz, false, false);
    cr = __builtin_amdgcn_wmma_f32_16x16x32_bf16(false, a, false, br, (short)0, cr, false, false);
    cn = __builtin_amdgcn_wmma_f32_16x16x32_bf16(false, a, false, bn, (short)0, cn, false, false);

    // ---- fused gates + h update ----
    float bn_add = bias[2 * Hn + ncol];

    // h_next packed position pieces (k = ncol fixed per lane)
    int pkt  = ncol >> 5;
    int pkin = ncol & 31;
    int phlf = (pkin >> 3) & 1;
    int pe   = (pkin & 7) + ((pkin >> 4) << 3);

    #pragma unroll
    for (int v = 0; v < 8; ++v) {
        int brow = m0 + mb + v;
        size_t g = ((size_t)brow * Sn + t) * Gn;
        float z   = sigmoidf_fast(gx[g + ncol] + cz[v]);
        float r   = sigmoidf_fast(gx[g + Hn + ncol] + cr[v]);
        float ghn = cn[v];
        float nn  = tanhf_fast(gx[g + 2 * Hn + ncol] + ghn + r * (ghn + bn_add));
        float hp  = hprevf[(size_t)brow * hstride + ncol];
        float hnew = (1.0f - z) * nn + z * hp;

        out[((size_t)brow * Sn + t) * Hn + ncol] = hnew;

        int prow = mb + v;                                  // row within m-tile
        int plane = phlf * 16 + prow;
        hpout[(((size_t)mt * 32 + pkt) * 32 + plane) * 16 + pe] = (__bf16)hnew;
    }
}

// ---------------------------------------------------------------------------
// Workspace layout (bytes):
//   [0,    6M)     W_ih packed bf16
//   [6M,   12M)    W_hh packed bf16
//   [12M,  12.25M) h packed buffer 0 (256KB)
//   [12.25M,12.5M) h packed buffer 1 (256KB)
//   [16M,  144M)   x bf16
//   [144M, 912M)   gx f32 (B,S,3H)
// ---------------------------------------------------------------------------
extern "C" void kernel_launch(void* const* d_in, const int* in_sizes, int n_in,
                              void* d_out, int out_size, void* d_ws, size_t ws_size,
                              hipStream_t stream) {
    const float* x    = (const float*)d_in[0];
    const float* Wih  = (const float*)d_in[1];
    const float* Whh  = (const float*)d_in[2];
    const float* bias = (const float*)d_in[3];
    const float* h0   = (const float*)d_in[4];
    float* out = (float*)d_out;

    char* ws = (char*)d_ws;
    __bf16* wihp = (__bf16*)(ws);
    __bf16* whhp = (__bf16*)(ws + (size_t)6 * 1024 * 1024);
    __bf16* hp0  = (__bf16*)(ws + (size_t)12 * 1024 * 1024);
    __bf16* hp1  = (__bf16*)(ws + (size_t)12 * 1024 * 1024 + 256 * 1024);
    __bf16* xb   = (__bf16*)(ws + (size_t)16 * 1024 * 1024);
    float*  gx   = (float*) (ws + (size_t)144 * 1024 * 1024);
    __bf16* hp[2] = { hp0, hp1 };

    // Pack weights into WMMA B-fragment order (bf16)
    pack_w_kernel<<<12288, 256, 0, stream>>>(Wih, wihp);   // 3072*1024 / 256
    pack_w_kernel<<<12288, 256, 0, stream>>>(Whh, whhp);

    // x f32 -> bf16 (float4 per thread)
    cvt_x_kernel<<<65536, 256, 0, stream>>>((const float4*)x, (v4bf*)xb);

    // hn output = h0 (B*H = 131072 elements) at d_out + B*S*H
    copy_hn_kernel<<<512, 256, 0, stream>>>(h0, out + (size_t)Bn * Sn * Hn);

    // h0 -> packed A-fragment layout (scan step 0 input)
    pack_h0_kernel<<<512, 256, 0, stream>>>(h0, hp[0]);

    // Phase A: gx = x @ W_ih^T + bias
    gemm_gx_kernel<<<dim3(2048, 12), 256, 0, stream>>>(xb, wihp, bias, gx);

    // Phase B: sequential scan; f32 h_{t-1} from out[:, t-1, :], bf16 A-frags
    // from the packed ping-pong buffers.
    for (int t = 0; t < Sn; ++t) {
        const float* hpf = (t == 0) ? h0 : (out + (size_t)(t - 1) * Hn);
        long hs = (t == 0) ? (long)Hn : (long)Sn * Hn;
        gru_step_kernel<<<dim3(Bn / 16, Hn / 64), 128, 0, stream>>>(
            t, gx, whhp, bias, hpf, hs,
            (const v16bf*)hp[t & 1], hp[(t + 1) & 1], out);
    }
}
